// MoEMLP_37933151158748
// MI455X (gfx1250) — matrix operand order repaired and checked by reading, not compiled, not run on hardware
//
#include <hip/hip_runtime.h>
#include <math.h>

#define DIMV 1024
#define EXP 8
#define HIDV 4096
#define NTOK 8192           // B*S = 4*2048
#define GATE_BLOCKS (NTOK / 8)

typedef __attribute__((ext_vector_type(16))) __bf16 v16bf;
typedef __attribute__((ext_vector_type(8)))  __bf16 v8bf;
typedef __attribute__((ext_vector_type(8)))  float  v8f;

#if __has_builtin(__builtin_amdgcn_global_load_async_to_lds_b128)
#define HAVE_ASYNC 1
#else
#define HAVE_ASYNC 0
#endif

typedef int v4i_t __attribute__((vector_size(16)));
typedef __attribute__((address_space(1))) v4i_t* gv4p_t;
typedef __attribute__((address_space(3))) v4i_t* lv4p_t;

__device__ __forceinline__ void copy16(const __bf16* g, __bf16* l) {
#if HAVE_ASYNC
    __builtin_amdgcn_global_load_async_to_lds_b128((gv4p_t)g, (lv4p_t)l, 0, 0);
#else
    *(v8bf*)l = *(const v8bf*)g;
#endif
}

__device__ __forceinline__ void wait_async() {
#if HAVE_ASYNC
#if __has_builtin(__builtin_amdgcn_s_wait_asynccnt)
    __builtin_amdgcn_s_wait_asynccnt(0);
#else
    asm volatile("s_wait_asynccnt 0x0" ::: "memory");
#endif
#endif
}

__device__ __forceinline__ v8bf zero8() {
    v8bf z;
    #pragma unroll
    for (int i = 0; i < 8; ++i) z[i] = (__bf16)0.0f;
    return z;
}

__device__ __forceinline__ v16bf load_frag(const __bf16* p0, const __bf16* p1) {
    v8bf a0 = *(const v8bf*)p0;
    v8bf a1 = *(const v8bf*)p1;
    return __builtin_shufflevector(a0, a1, 0,1,2,3,4,5,6,7,8,9,10,11,12,13,14,15);
}

// ---------------------------------------------------------------------------
// x (fp32) -> bf16 straight convert. 8 elements / thread.
// ---------------------------------------------------------------------------
__global__ __launch_bounds__(256) void moe_convx_kernel(
    const float* __restrict__ x, __bf16* __restrict__ xbf)
{
    size_t i = ((size_t)blockIdx.x * 256 + threadIdx.x) * 8;
    float4 f0 = *(const float4*)(x + i);
    float4 f1 = *(const float4*)(x + i + 4);
    v8bf v;
    v[0] = (__bf16)f0.x; v[1] = (__bf16)f0.y; v[2] = (__bf16)f0.z; v[3] = (__bf16)f0.w;
    v[4] = (__bf16)f1.x; v[5] = (__bf16)f1.y; v[6] = (__bf16)f1.z; v[7] = (__bf16)f1.w;
    *(v8bf*)(xbf + i) = v;
}

// ---------------------------------------------------------------------------
// Weight convert + transpose: W[e][K][N] fp32 -> Wt[e][N][K] bf16.
// ---------------------------------------------------------------------------
__global__ __launch_bounds__(256) void moe_convw_kernel(
    const float* __restrict__ W, __bf16* __restrict__ Wt, int K, int N)
{
    __shared__ __align__(16) __bf16 T[64][72];   // T[n_local][k_local]
    int e  = blockIdx.z;
    int n0 = blockIdx.x * 64;
    int k0 = blockIdx.y * 64;
    int tid = threadIdx.x;
    {
        int flat = tid * 16;
        int kk = flat >> 6;
        int nn = flat & 63;
        const float* src = W + ((size_t)e * K + k0 + kk) * N + n0 + nn;
        #pragma unroll
        for (int i = 0; i < 16; ++i) T[nn + i][kk] = (__bf16)src[i];
    }
    __syncthreads();
    {
        int flat = tid * 16;
        int nn = flat >> 6;
        int kk = flat & 63;
        __bf16* dst = Wt + ((size_t)e * N + n0 + nn) * K + k0 + kk;
        *(v8bf*)(dst)     = *(const v8bf*)&T[nn][kk];
        *(v8bf*)(dst + 8) = *(const v8bf*)&T[nn][kk + 8];
    }
}

// ---------------------------------------------------------------------------
// Gating: one wave32 per token, deterministic per-block partial sums.
// ---------------------------------------------------------------------------
__global__ __launch_bounds__(256) void moe_gate_kernel(
    const float* __restrict__ x, const float* __restrict__ gW,
    const float* __restrict__ gb, int* __restrict__ idx,
    float* __restrict__ score, float* __restrict__ blkProbs,
    int* __restrict__ blkCounts)
{
    int wave = threadIdx.x >> 5;
    int lane = threadIdx.x & 31;
    int tok  = blockIdx.x * 8 + wave;

    float acc[EXP];
    #pragma unroll
    for (int e = 0; e < EXP; ++e) acc[e] = 0.f;

    const float* xr = x + (size_t)tok * DIMV;
    for (int d = lane; d < DIMV; d += 32) {
        float xv = xr[d];
        const float* g = gW + (size_t)d * EXP;
        #pragma unroll
        for (int e = 0; e < EXP; ++e) acc[e] += xv * g[e];
    }
    #pragma unroll
    for (int e = 0; e < EXP; ++e) {
        float v = acc[e];
        #pragma unroll
        for (int off = 16; off > 0; off >>= 1) v += __shfl_xor(v, off, 32);
        acc[e] = v;
    }

    __shared__ float sprob[8][EXP];
    __shared__ int   scnt[8][EXP];
    if (lane == 0) {
        float logits[EXP];
        float m = -1e30f; int mi = 0;
        #pragma unroll
        for (int e = 0; e < EXP; ++e) {
            logits[e] = acc[e] + gb[e];
            if (logits[e] > m) { m = logits[e]; mi = e; }
        }
        float p[EXP]; float se = 0.f;
        #pragma unroll
        for (int e = 0; e < EXP; ++e) { p[e] = expf(logits[e] - m); se += p[e]; }
        float inv = 1.f / se;
        #pragma unroll
        for (int e = 0; e < EXP; ++e) {
            p[e] *= inv;
            sprob[wave][e] = p[e];
            scnt[wave][e]  = (e == mi) ? 1 : 0;
        }
        idx[tok]   = mi;
        score[tok] = p[mi];
    }
    __syncthreads();
    if (threadIdx.x == 0) {
        #pragma unroll
        for (int e = 0; e < EXP; ++e) {
            float s = 0.f; int c = 0;
            for (int w = 0; w < 8; ++w) { s += sprob[w][e]; c += scnt[w][e]; }
            blkProbs[blockIdx.x * EXP + e]  = s;
            blkCounts[blockIdx.x * EXP + e] = c;
        }
    }
}

__global__ void moe_finalize_kernel(
    const float* __restrict__ blkProbs, const int* __restrict__ blkCounts,
    int* __restrict__ counts, int* __restrict__ offsets, int* __restrict__ cursor,
    float* __restrict__ out_loss, int nblk)
{
    if (threadIdx.x != 0 || blockIdx.x != 0) return;
    float mg[EXP]; int cnt[EXP];
    for (int e = 0; e < EXP; ++e) { mg[e] = 0.f; cnt[e] = 0; }
    for (int b = 0; b < nblk; ++b)
        for (int e = 0; e < EXP; ++e) {
            mg[e]  += blkProbs[b * EXP + e];
            cnt[e] += blkCounts[b * EXP + e];
        }
    float loss = 0.f;
    for (int e = 0; e < EXP; ++e)
        loss += (mg[e] / (float)NTOK) * ((float)cnt[e] / (float)NTOK);
    *out_loss = loss * (float)EXP;
    int off = 0;
    for (int e = 0; e < EXP; ++e) {
        offsets[e] = off; counts[e] = cnt[e]; cursor[e] = 0; off += cnt[e];
    }
}

__global__ __launch_bounds__(256) void moe_scatter_kernel(
    const int* __restrict__ idx, const int* __restrict__ offsets,
    int* __restrict__ cursor, int* __restrict__ perm)
{
    int t = blockIdx.x * 256 + threadIdx.x;
    int e = idx[t];
    int pos = atomicAdd(&cursor[e], 1);
    perm[offsets[e] + pos] = t;
}

// ---------------------------------------------------------------------------
// Double-buffered WMMA GEMM: 256 threads = 8 waves, block tile 64(M)x256(N),
// K-stage 32, pipeline unrolled x2 so each copy targets a constant buffer.
// ---------------------------------------------------------------------------
#define WMMA_BF16(A, Bv, C) __builtin_amdgcn_wmma_f32_16x16x32_bf16( \
    false, (A), false, (Bv), (short)0, (C), false, false)

#define COMPUTE_STAGE(b) do { \
    v16bf af0 = load_frag(&As[b][wy + lr][half * 8], \
                          &As[b][wy + lr][16 + half * 8]); \
    v16bf af1 = load_frag(&As[b][wy + 16 + lr][half * 8], \
                          &As[b][wy + 16 + lr][16 + half * 8]); \
    v16bf bv[4]; \
    _Pragma("unroll") \
    for (int nj = 0; nj < 4; ++nj) \
        bv[nj] = load_frag(&Bs[b][wx + nj * 16 + lr][half * 8], \
                           &Bs[b][wx + nj * 16 + lr][16 + half * 8]); \
    _Pragma("unroll") \
    for (int nj = 0; nj < 4; ++nj) { \
        acc[0][nj] = WMMA_BF16(af0, bv[nj], acc[0][nj]); \
        acc[1][nj] = WMMA_BF16(af1, bv[nj], acc[1][nj]); \
    } } while (0)

// GEMM1: h = gelu(x_gathered @ W1_e + b1_e)
__global__ __launch_bounds__(256) void moe_gemm1_kernel(
    const __bf16* __restrict__ xbf, const __bf16* __restrict__ W1t,
    const float* __restrict__ b1, const int* __restrict__ counts,
    const int* __restrict__ offsets, const int* __restrict__ perm,
    __bf16* __restrict__ h)
{
    int e   = blockIdx.z;
    int cnt = counts[e];
    int m0  = blockIdx.y * 64;
    if (m0 >= cnt) return;
    int n0  = blockIdx.x * 256;
    int off = offsets[e];
    const __bf16* Wt = W1t + (size_t)e * HIDV * DIMV;

    __shared__ __align__(16) __bf16 As[2][64][40];
    __shared__ __align__(16) __bf16 Bs[2][256][40];
    __shared__ int rowtok[64];

    int tid = threadIdx.x;
    if (tid < 64) {
        int m = m0 + tid;
        rowtok[tid] = (m < cnt) ? perm[off + m] : -1;
    }
    __syncthreads();

    int ar = tid >> 2, apos = (tid & 3) * 8;          // 64 rows x 4 chunks
    int atok = rowtok[ar];
    const __bf16* asrc = xbf + (size_t)(atok < 0 ? 0 : atok) * DIMV + apos;
    __bf16* adst0 = &As[0][ar][apos];
    __bf16* adst1 = &As[1][ar][apos];
    bool aok = (atok >= 0);
    if (!aok) { *(v8bf*)adst0 = zero8(); *(v8bf*)adst1 = zero8(); }

    const __bf16* bsrc[4]; __bf16* bdst0[4]; __bf16* bdst1[4];
    #pragma unroll
    for (int i = 0; i < 4; ++i) {
        int c = i * 256 + tid;                         // 1024 chunks: 256 rows x 4
        int r = c >> 2, pos = (c & 3) * 8;
        bsrc[i]  = Wt + (size_t)(n0 + r) * DIMV + pos;
        bdst0[i] = &Bs[0][r][pos];
        bdst1[i] = &Bs[1][r][pos];
    }

    int wave = tid >> 5, lane = tid & 31;
    int wy = (wave & 1) * 32, wx = (wave >> 1) * 64;
    int half = lane >> 4, lr = lane & 15;

    v8f acc[2][4];
    #pragma unroll
    for (int mi = 0; mi < 2; ++mi)
        #pragma unroll
        for (int nj = 0; nj < 4; ++nj) acc[mi][nj] = v8f{};

    // prologue: stage 0 -> buffer 0
    if (aok) copy16(asrc, adst0);
    #pragma unroll
    for (int i = 0; i < 4; ++i) copy16(bsrc[i], bdst0[i]);

    const int NSTAGE = DIMV / 32;                     // even
    for (int s = 0; s < NSTAGE; s += 2) {
        wait_async();
        __syncthreads();
        {   // stage s+1 -> buffer 1 (always exists: NSTAGE even)
            int kn = (s + 1) * 32;
            if (aok) copy16(asrc + kn, adst1);
            #pragma unroll
            for (int i = 0; i < 4; ++i) copy16(bsrc[i] + kn, bdst1[i]);
        }
        COMPUTE_STAGE(0);
        wait_async();
        __syncthreads();
        if (s + 2 < NSTAGE) {   // stage s+2 -> buffer 0
            int kn = (s + 2) * 32;
            if (aok) copy16(asrc + kn, adst0);
            #pragma unroll
            for (int i = 0; i < 4; ++i) copy16(bsrc[i] + kn, bdst0[i]);
        }
        COMPUTE_STAGE(1);
    }

    #pragma unroll
    for (int mi = 0; mi < 2; ++mi)
        #pragma unroll
        for (int nj = 0; nj < 4; ++nj) {
            int col = n0 + wx + nj * 16 + lr;
            float bias = b1[(size_t)e * HIDV + col];
            #pragma unroll
            for (int i = 0; i < 8; ++i) {
                int m = m0 + wy + mi * 16 + i + 8 * half;
                if (m < cnt) {
                    float v = acc[mi][nj][i] + bias;
                    float g = 0.5f * v * (1.0f + erff(v * 0.70710678118654752f));
                    h[(size_t)(off + m) * HIDV + col] = (__bf16)g;
                }
            }
        }
}

// GEMM2: out[token] = (h_row @ W2_e + b2_e) * score[token]
__global__ __launch_bounds__(256) void moe_gemm2_kernel(
    const __bf16* __restrict__ h, const __bf16* __restrict__ W2t,
    const float* __restrict__ b2, const int* __restrict__ counts,
    const int* __restrict__ offsets, const int* __restrict__ perm,
    const float* __restrict__ score, float* __restrict__ out)
{
    int e   = blockIdx.z;
    int cnt = counts[e];
    int m0  = blockIdx.y * 64;
    if (m0 >= cnt) return;
    int n0  = blockIdx.x * 256;
    int off = offsets[e];
    const __bf16* Wt = W2t + (size_t)e * DIMV * HIDV;

    __shared__ __align__(16) __bf16 As[2][64][40];
    __shared__ __align__(16) __bf16 Bs[2][256][40];
    __shared__ int rowtok[64];

    int tid = threadIdx.x;
    if (tid < 64) {
        int m = m0 + tid;
        rowtok[tid] = (m < cnt) ? perm[off + m] : -1;
    }
    __syncthreads();

    int ar = tid >> 2, apos = (tid & 3) * 8;
    bool aok = (m0 + ar) < cnt;
    const __bf16* asrc = h + (size_t)(off + (aok ? (m0 + ar) : 0)) * HIDV + apos;
    __bf16* adst0 = &As[0][ar][apos];
    __bf16* adst1 = &As[1][ar][apos];
    if (!aok) { *(v8bf*)adst0 = zero8(); *(v8bf*)adst1 = zero8(); }

    const __bf16* bsrc[4]; __bf16* bdst0[4]; __bf16* bdst1[4];
    #pragma unroll
    for (int i = 0; i < 4; ++i) {
        int c = i * 256 + tid;
        int r = c >> 2, pos = (c & 3) * 8;
        bsrc[i]  = Wt + (size_t)(n0 + r) * HIDV + pos;
        bdst0[i] = &Bs[0][r][pos];
        bdst1[i] = &Bs[1][r][pos];
    }

    int wave = tid >> 5, lane = tid & 31;
    int wy = (wave & 1) * 32, wx = (wave >> 1) * 64;
    int half = lane >> 4, lr = lane & 15;

    v8f acc[2][4];
    #pragma unroll
    for (int mi = 0; mi < 2; ++mi)
        #pragma unroll
        for (int nj = 0; nj < 4; ++nj) acc[mi][nj] = v8f{};

    if (aok) copy16(asrc, adst0);
    #pragma unroll
    for (int i = 0; i < 4; ++i) copy16(bsrc[i], bdst0[i]);

    const int NSTAGE = HIDV / 32;                     // even
    for (int s = 0; s < NSTAGE; s += 2) {
        wait_async();
        __syncthreads();
        {
            int kn = (s + 1) * 32;
            if (aok) copy16(asrc + kn, adst1);
            #pragma unroll
            for (int i = 0; i < 4; ++i) copy16(bsrc[i] + kn, bdst1[i]);
        }
        COMPUTE_STAGE(0);
        wait_async();
        __syncthreads();
        if (s + 2 < NSTAGE) {
            int kn = (s + 2) * 32;
            if (aok) copy16(asrc + kn, adst0);
            #pragma unroll
            for (int i = 0; i < 4; ++i) copy16(bsrc[i] + kn, bdst0[i]);
        }
        COMPUTE_STAGE(1);
    }

    #pragma unroll
    for (int mi = 0; mi < 2; ++mi)
        #pragma unroll
        for (int nj = 0; nj < 4; ++nj) {
            int col = n0 + wx + nj * 16 + lr;
            float bias = b2[(size_t)e * DIMV + col];
            #pragma unroll
            for (int i = 0; i < 8; ++i) {
                int m = m0 + wy + mi * 16 + i + 8 * half;
                if (m < cnt) {
                    int t = rowtok[wy + mi * 16 + i + 8 * half];
                    out[(size_t)t * DIMV + col] = (acc[mi][nj][i] + bias) * score[t];
                }
            }
        }
}

// ---------------------------------------------------------------------------
extern "C" void kernel_launch(void* const* d_in, const int* in_sizes, int n_in,
                              void* d_out, int out_size, void* d_ws, size_t ws_size,
                              hipStream_t stream)
{
    const float* x  = (const float*)d_in[0];
    const float* gW = (const float*)d_in[1];
    const float* gb = (const float*)d_in[2];
    const float* W1 = (const float*)d_in[3];
    const float* b1 = (const float*)d_in[4];
    const float* W2 = (const float*)d_in[5];
    const float* b2 = (const float*)d_in[6];
    float* out = (float*)d_out;

    char* ws = (char*)d_ws;
    size_t p = 0;
    auto alloc = [&](size_t bytes) -> char* {
        char* r = ws + p;
        p += (bytes + 255) & ~(size_t)255;
        return r;
    };
    __bf16* h    = (__bf16*)alloc((size_t)NTOK * HIDV * sizeof(__bf16));
    __bf16* xbf  = (__bf16*)alloc((size_t)NTOK * DIMV * sizeof(__bf16));
    __bf16* wt   = (__bf16*)alloc((size_t)EXP * HIDV * DIMV * sizeof(__bf16)); // W1t then W2t
    float*  blkProbs = (float*)alloc((size_t)GATE_BLOCKS * EXP * sizeof(float));
    int*    blkCnts  = (int*)alloc((size_t)GATE_BLOCKS * EXP * sizeof(int));
    int*    counts   = (int*)alloc(EXP * sizeof(int));
    int*    offsets  = (int*)alloc(EXP * sizeof(int));
    int*    cursor   = (int*)alloc(EXP * sizeof(int));
    int*    idx      = (int*)alloc(NTOK * sizeof(int));
    float*  score    = (float*)alloc(NTOK * sizeof(float));
    int*    perm     = (int*)alloc(NTOK * sizeof(int));
    (void)ws_size; (void)in_sizes; (void)n_in; (void)out_size;

    float* out_loss = out + (size_t)NTOK * DIMV;

    moe_convx_kernel<<<(NTOK * DIMV) / (256 * 8), 256, 0, stream>>>(x, xbf);
    moe_convw_kernel<<<dim3(HIDV / 64, DIMV / 64, EXP), 256, 0, stream>>>(
        W1, wt, DIMV, HIDV);
    moe_gate_kernel<<<GATE_BLOCKS, 256, 0, stream>>>(x, gW, gb, idx, score,
                                                     blkProbs, blkCnts);
    moe_finalize_kernel<<<1, 32, 0, stream>>>(blkProbs, blkCnts, counts, offsets,
                                              cursor, out_loss, GATE_BLOCKS);
    moe_scatter_kernel<<<NTOK / 256, 256, 0, stream>>>(idx, offsets, cursor, perm);
    moe_gemm1_kernel<<<dim3(HIDV / 256, NTOK / 64, EXP), 256, 0, stream>>>(
        xbf, wt, b1, counts, offsets, perm, h);
    moe_convw_kernel<<<dim3(DIMV / 64, HIDV / 64, EXP), 256, 0, stream>>>(
        W2, wt, HIDV, DIMV);
    moe_gemm2_kernel<<<dim3(DIMV / 256, NTOK / 64, EXP), 256, 0, stream>>>(
        h, wt, b2, counts, offsets, perm, score, out);
}
